// SpatialSelfAttention2d_13778255085771
// MI455X (gfx1250) — matrix operand order repaired and checked by reading, not compiled, not run on hardware
//
#include <hip/hip_runtime.h>
#include <hip/hip_bf16.h>

typedef __attribute__((ext_vector_type(16))) _Float16 v16h;
typedef __attribute__((ext_vector_type(8)))  _Float16 v8h;
typedef __attribute__((ext_vector_type(4)))  _Float16 v4h;
typedef __attribute__((ext_vector_type(8)))  float    v8f;

typedef __attribute__((ext_vector_type(4))) unsigned int tdm_g0_t;
typedef __attribute__((ext_vector_type(8))) int          tdm_g1_t;
typedef __attribute__((ext_vector_type(4))) int          tdm_g2_t;
typedef __attribute__((ext_vector_type(8))) int          tdm_g4_t;

#define CC 256
#define NN 4096
#define NH 4
#define HD 64

__device__ __forceinline__ v8f zero8() {
  v8f z;
#pragma unroll
  for (int i = 0; i < 8; ++i) z[i] = 0.0f;
  return z;
}

__device__ __forceinline__ v8f wmma16(v16h a, v16h b, v8f c) {
  // D = A(16x32 f16) * B(32x16 f16) + C(16x16 f32)
  return __builtin_amdgcn_wmma_f32_16x16x32_f16(
      /*neg_a=*/false, a, /*neg_b=*/false, b,
      /*c_mod=*/(short)0, c, /*reuse_a=*/false, /*reuse_b=*/false);
}

// A-fragment (16x32 f16, row-major source): lane holds row m = lm, with
// K elements j -> k = hseg*16 + j
__device__ __forceinline__ v16h load_a_f16(const _Float16* rowp, int hseg) {
  v8h c0 = *(const v8h*)(rowp + hseg * 16);
  v8h c1 = *(const v8h*)(rowp + hseg * 16 + 8);
  v16h a;
#pragma unroll
  for (int i = 0; i < 8; ++i) { a[i] = c0[i]; a[8 + i] = c1[i]; }
  return a;
}

// --- Tensor Data Mover: 2D tile load, D# per CDNA5 ISA.
// Tensor dims are set equal to tile dims so the OOB path can never trigger.
__device__ __forceinline__ unsigned lds_off32(const void* p) {
  return (unsigned)(uintptr_t)p;  // low 32 bits of flat LDS addr = LDS offset
}

__device__ __forceinline__ void tdm_load_2d(unsigned lds_off, const void* gptr,
                                            unsigned tile0, unsigned tile1,
                                            unsigned stride0 /* elements */) {
  const unsigned long long ga = (unsigned long long)(uintptr_t)gptr;
  tdm_g0_t g0;
  g0[0] = 1u;                                   // count=1, user descriptor
  g0[1] = lds_off;                              // lds_addr (bytes)
  g0[2] = (unsigned)ga;                         // global_addr[31:0]
  g0[3] = (unsigned)((ga >> 32) & 0x1FFFFFFu)   // global_addr[56:32]
          | 0x80000000u;                        // type=2 ("image")
  tdm_g1_t g1;
  g1[0] = (int)(1u << 16);                      // data_size=1 -> 2 bytes
  g1[1] = (int)((tile0 & 0xFFFFu) << 16);       // tensor_dim0[15:0]
  g1[2] = (int)((tile0 >> 16) | ((tile1 & 0xFFFFu) << 16));  // dim0 hi | dim1 lo
  g1[3] = (int)((tile1 >> 16) | (tile0 << 16)); // dim1 hi | tile_dim0
  g1[4] = (int)tile1;                           // tile_dim1 (tile_dim2=0)
  g1[5] = (int)stride0;                         // tensor_dim0_stride[31:0]
  g1[6] = 0;                                    // stride0 hi | stride1 lo
  g1[7] = 0;                                    // stride1 hi
  tdm_g2_t z;
  z[0] = z[1] = z[2] = z[3] = 0;
  tdm_g4_t z8;
#pragma unroll
  for (int i = 0; i < 8; ++i) z8[i] = 0;
  __builtin_amdgcn_tensor_load_to_lds(g0, g1, z, z, z8, 0);
}

// ---------------------------------------------------------------------------
// Kernel 0: one-time f32 -> f16 weight conversion (keeps cvt out of GEMM loop)
// ---------------------------------------------------------------------------
__global__ void cvt_f32_to_f16(const float* __restrict__ src,
                               _Float16* __restrict__ dst, int n) {
  const int i = (blockIdx.x * blockDim.x + threadIdx.x) * 4;
  if (i + 3 < n) {
    const float4 v = *(const float4*)(src + i);
    v4h o;
    o[0] = (_Float16)v.x; o[1] = (_Float16)v.y;
    o[2] = (_Float16)v.z; o[3] = (_Float16)v.w;
    *(v4h*)(dst + i) = o;
  }
}

// ---------------------------------------------------------------------------
// Kernel 1: GroupNorm (32 groups of 8 channels over HxW) -> f16, transposed
// to hT[B][N][C] so GEMM B-fragments are contiguous over C.
// ---------------------------------------------------------------------------
__global__ void groupnorm_to_f16T(const float* __restrict__ x,
                                  const float* __restrict__ gamma,
                                  const float* __restrict__ beta,
                                  _Float16* __restrict__ hT) {
  const int b = blockIdx.x >> 5;
  const int g = blockIdx.x & 31;
  const int tid = threadIdx.x;
  const float* xb = x + ((size_t)b * CC + g * 8) * NN;

  float s = 0.f, ss = 0.f;
  for (int n = tid; n < NN; n += 256) {
#pragma unroll
    for (int c = 0; c < 8; ++c) {
      float v = xb[(size_t)c * NN + n];
      s += v; ss += v * v;
    }
  }
#pragma unroll
  for (int m = 16; m >= 1; m >>= 1) {
    s  += __shfl_xor(s,  m, 32);
    ss += __shfl_xor(ss, m, 32);
  }
  __shared__ float reds[8], redss[8];
  const int wid = tid >> 5;
  if ((tid & 31) == 0) { reds[wid] = s; redss[wid] = ss; }
  __syncthreads();
  s = 0.f; ss = 0.f;
#pragma unroll
  for (int w = 0; w < 8; ++w) { s += reds[w]; ss += redss[w]; }

  const float inv_cnt = 1.0f / (8.0f * (float)NN);
  const float mu = s * inv_cnt;
  const float rstd = rsqrtf(ss * inv_cnt - mu * mu + 1e-5f);

  float gm[8], bt[8];
#pragma unroll
  for (int c = 0; c < 8; ++c) {
    gm[c] = gamma[g * 8 + c] * rstd;
    bt[c] = beta[g * 8 + c] - mu * gm[c];
  }
  for (int n = tid; n < NN; n += 256) {
    v8h o;
#pragma unroll
    for (int c = 0; c < 8; ++c)
      o[c] = (_Float16)(xb[(size_t)c * NN + n] * gm[c] + bt[c]);
    *(v8h*)(hT + ((size_t)b * NN + n) * CC + g * 8) = o;  // 16B store
  }
}

// ---------------------------------------------------------------------------
// Kernel 2: QKV GEMM  out[o,n] = sum_c W[o,c]*h[c,n] + b[o]  (M=768,N=4096,K=256)
// Block = 8 waves, block tile 128x128, wave tile 32x64 (2x4 of 16x16).
// Scatters to Q,K [B,nh,N,hd] (Q pre-scaled by hd^-0.5) and V transposed
// Vt [B,nh,hd,N].
// ---------------------------------------------------------------------------
__global__ void qkv_gemm(const _Float16* __restrict__ hT,
                         const _Float16* __restrict__ w16,
                         const float* __restrict__ qkv_b,
                         _Float16* __restrict__ Q,
                         _Float16* __restrict__ K,
                         _Float16* __restrict__ Vt) {
  const int b    = blockIdx.z;
  const int wave = threadIdx.x >> 5;
  const int lane = threadIdx.x & 31;
  const int hseg = lane >> 4, lm = lane & 15;
  const int m0 = blockIdx.y * 128 + (wave & 3) * 32;
  const int n0 = blockIdx.x * 128 + (wave >> 2) * 64;
  const _Float16* hb = hT + (size_t)b * NN * CC;

  v8f acc[2][4];
#pragma unroll
  for (int mt = 0; mt < 2; ++mt)
#pragma unroll
    for (int nt = 0; nt < 4; ++nt) acc[mt][nt] = zero8();

  for (int k0 = 0; k0 < CC; k0 += 32) {
    v16h a[2], bm[4];
#pragma unroll
    for (int mt = 0; mt < 2; ++mt)
      a[mt] = load_a_f16(w16 + (size_t)(m0 + mt * 16 + lm) * CC + k0, hseg);
#pragma unroll
    for (int nt = 0; nt < 4; ++nt)   // B-fragment: lane=col n, 16 contiguous c
      bm[nt] = *(const v16h*)(hb + (size_t)(n0 + nt * 16 + lm) * CC + k0 + hseg * 16);
#pragma unroll
    for (int mt = 0; mt < 2; ++mt)
#pragma unroll
      for (int nt = 0; nt < 4; ++nt)
        acc[mt][nt] = wmma16(a[mt], bm[nt], acc[mt][nt]);
    if (k0 + 32 < CC)
      __builtin_prefetch(hb + (size_t)(n0 + lm) * CC + k0 + 32, 0, 1);
  }

  const int sec = m0 >> 8;  // 0=q, 1=k, 2=v (uniform per wave: 32-row strip)
#pragma unroll
  for (int mt = 0; mt < 2; ++mt) {
#pragma unroll
    for (int nt = 0; nt < 4; ++nt) {
#pragma unroll
      for (int r = 0; r < 8; ++r) {
        const int o  = m0 + mt * 16 + r + 8 * hseg;  // C-layout row
        const int n  = n0 + nt * 16 + lm;            // C-layout col
        const int ch = o & 255, head = ch >> 6, d = ch & 63;
        const float v = acc[mt][nt][r] + qkv_b[o];
        if (sec == 0)
          Q[(((size_t)b * NH + head) * NN + n) * HD + d] = (_Float16)(v * 0.125f);
        else if (sec == 1)
          K[(((size_t)b * NH + head) * NN + n) * HD + d] = (_Float16)v;
        else
          Vt[(((size_t)b * NH + head) * HD + d) * NN + n] = (_Float16)v;
      }
    }
  }
}

// ---------------------------------------------------------------------------
// Kernel 3: Flash attention. grid=(N/64, B*nh), block=128 (4 waves).
// K/V tiles (32 keys) are staged into LDS by the Tensor Data Mover, double
// buffered, tracked with TENSORcnt; all 4 waves read B-fragments from LDS.
//   S = Q K^T (4 wmma) -> online softmax (half-wave shfl row reductions)
//   P transposed C-layout -> A-layout via per-wave LDS tile
//   O += P V (4 wmma).  Writes attnO[B][N][C] f16 (input to proj GEMM).
// ---------------------------------------------------------------------------
__global__ void flash_attn(const _Float16* __restrict__ Q,
                           const _Float16* __restrict__ K,
                           const _Float16* __restrict__ Vt,
                           _Float16* __restrict__ attnO) {
  const int bh   = blockIdx.y;
  const int wave = threadIdx.x >> 5;
  const int lane = threadIdx.x & 31;
  const int hseg = lane >> 4, lm = lane & 15;
  const int row0 = blockIdx.x * 64 + wave * 16;

  const _Float16* Qb = Q  + (size_t)bh * NN * HD;
  const _Float16* Kb = K  + (size_t)bh * NN * HD;
  const _Float16* Vb = Vt + (size_t)bh * HD * NN;

  __shared__ _Float16 klds[2][32 * HD];   // [key][d]   4KB per buffer
  __shared__ _Float16 vlds[2][HD * 32];   // [d][key]   4KB per buffer
  __shared__ _Float16 plds[4][16 * 48];   // per-wave P tile, 96B rows
  _Float16* myp = plds[wave];

  v16h qa[2];
#pragma unroll
  for (int ks = 0; ks < 2; ++ks)
    qa[ks] = load_a_f16(Qb + (size_t)(row0 + lm) * HD + ks * 32, hseg);

  float mrow[8], lrow[8];
  v8f o[4];
#pragma unroll
  for (int r = 0; r < 8; ++r) { mrow[r] = -1e30f; lrow[r] = 0.f; }
#pragma unroll
  for (int t = 0; t < 4; ++t) o[t] = zero8();

  // Prologue: TDM-load tile 0 (wave 0 issues, all waves sync on completion).
  if (wave == 0) {
    tdm_load_2d(lds_off32(&klds[0][0]), Kb, HD, 32, HD);
    tdm_load_2d(lds_off32(&vlds[0][0]), Vb, 32, HD, NN);
    __builtin_amdgcn_s_wait_tensorcnt(0);
  }
  __syncthreads();

  const int ntile = NN / 32;
  for (int it = 0; it < ntile; ++it) {
    const int cur = it & 1;
    // Async prefetch of tile it+1 into the other buffer (freed by the
    // barrier that ended iteration it-1).
    if (wave == 0 && it + 1 < ntile) {
      const int c1 = (it + 1) * 32;
      tdm_load_2d(lds_off32(&klds[cur ^ 1][0]), Kb + (size_t)c1 * HD, HD, 32, HD);
      tdm_load_2d(lds_off32(&vlds[cur ^ 1][0]), Vb + c1, 32, HD, NN);
    }

    v8f s[2];
    s[0] = zero8(); s[1] = zero8();
#pragma unroll
    for (int nt = 0; nt < 2; ++nt)
#pragma unroll
      for (int ks = 0; ks < 2; ++ks) {
        // B-frag of K^T from LDS: lane=key col, 16 contiguous d values
        v16h kb = *(const v16h*)(&klds[cur][(nt * 16 + lm) * HD + ks * 32 + hseg * 16]);
        s[nt] = wmma16(qa[ks], kb, s[nt]);
      }

    // ---- online softmax: C-layout => rows r+8*hseg, cols across 16 lanes ----
    float alpha[8];
#pragma unroll
    for (int r = 0; r < 8; ++r) {
      float mx = fmaxf(s[0][r], s[1][r]);
#pragma unroll
      for (int msk = 8; msk >= 1; msk >>= 1)
        mx = fmaxf(mx, __shfl_xor(mx, msk, 32));
      const float mnew = fmaxf(mrow[r], mx);
      alpha[r] = __expf(mrow[r] - mnew);
      mrow[r] = mnew;
    }
#pragma unroll
    for (int nt = 0; nt < 2; ++nt)
#pragma unroll
      for (int r = 0; r < 8; ++r)
        s[nt][r] = __expf(s[nt][r] - mrow[r]);
#pragma unroll
    for (int r = 0; r < 8; ++r) {
      float ps = s[0][r] + s[1][r];
#pragma unroll
      for (int msk = 8; msk >= 1; msk >>= 1)
        ps += __shfl_xor(ps, msk, 32);
      lrow[r] = lrow[r] * alpha[r] + ps;
    }
#pragma unroll
    for (int t = 0; t < 4; ++t)
#pragma unroll
      for (int r = 0; r < 8; ++r)
        o[t][r] *= alpha[r];

    // ---- transpose P (C-layout -> A-layout) through wave-private LDS ----
#pragma unroll
    for (int nt = 0; nt < 2; ++nt)
#pragma unroll
      for (int r = 0; r < 8; ++r)
        myp[(r + 8 * hseg) * 48 + nt * 16 + lm] = (_Float16)s[nt][r];

    v8h p0 = *(const v8h*)(&myp[lm * 48 + hseg * 16]);
    v8h p1 = *(const v8h*)(&myp[lm * 48 + hseg * 16 + 8]);
    v16h pa;
#pragma unroll
    for (int i = 0; i < 8; ++i) { pa[i] = p0[i]; pa[8 + i] = p1[i]; }

    // ---- O += P * V : B-frag from LDS [d][key] tile, contiguous over keys --
#pragma unroll
    for (int t = 0; t < 4; ++t) {
      v16h vb = *(const v16h*)(&vlds[cur][(t * 16 + lm) * 32 + hseg * 16]);
      o[t] = wmma16(pa, vb, o[t]);
    }

    __syncthreads();  // all waves done reading klds/vlds[cur]
    if (wave == 0) __builtin_amdgcn_s_wait_tensorcnt(0);  // tile it+1 landed
    __syncthreads();  // publish: next buffer valid for everyone
  }

  const int b = bh >> 2, head = bh & 3;
  float inv[8];
#pragma unroll
  for (int r = 0; r < 8; ++r) inv[r] = 1.0f / lrow[r];
#pragma unroll
  for (int t = 0; t < 4; ++t)
#pragma unroll
    for (int r = 0; r < 8; ++r) {
      const int row = row0 + r + 8 * hseg;
      const int col = head * HD + t * 16 + lm;
      attnO[((size_t)b * NN + row) * CC + col] = (_Float16)(o[t][r] * inv[r]);
    }
}

// ---------------------------------------------------------------------------
// Kernel 4: proj GEMM (M=256,N=4096,K=256) + bias + residual, f32 out [B,C,H,W]
// ---------------------------------------------------------------------------
__global__ void proj_gemm(const _Float16* __restrict__ attnO,
                          const _Float16* __restrict__ w16,
                          const float* __restrict__ proj_b,
                          const float* __restrict__ x,
                          float* __restrict__ out) {
  const int b    = blockIdx.z;
  const int wave = threadIdx.x >> 5;
  const int lane = threadIdx.x & 31;
  const int hseg = lane >> 4, lm = lane & 15;
  const int m0 = blockIdx.y * 128 + (wave & 3) * 32;
  const int n0 = blockIdx.x * 128 + (wave >> 2) * 64;
  const _Float16* ab = attnO + (size_t)b * NN * CC;

  v8f acc[2][4];
#pragma unroll
  for (int mt = 0; mt < 2; ++mt)
#pragma unroll
    for (int nt = 0; nt < 4; ++nt) acc[mt][nt] = zero8();

  for (int k0 = 0; k0 < CC; k0 += 32) {
    v16h a[2], bm[4];
#pragma unroll
    for (int mt = 0; mt < 2; ++mt)
      a[mt] = load_a_f16(w16 + (size_t)(m0 + mt * 16 + lm) * CC + k0, hseg);
#pragma unroll
    for (int nt = 0; nt < 4; ++nt)
      bm[nt] = *(const v16h*)(ab + (size_t)(n0 + nt * 16 + lm) * CC + k0 + hseg * 16);
#pragma unroll
    for (int mt = 0; mt < 2; ++mt)
#pragma unroll
      for (int nt = 0; nt < 4; ++nt)
        acc[mt][nt] = wmma16(a[mt], bm[nt], acc[mt][nt]);
    if (k0 + 32 < CC)
      __builtin_prefetch(ab + (size_t)(n0 + lm) * CC + k0 + 32, 0, 1);
  }

#pragma unroll
  for (int mt = 0; mt < 2; ++mt)
#pragma unroll
    for (int nt = 0; nt < 4; ++nt)
#pragma unroll
      for (int r = 0; r < 8; ++r) {
        const int o = m0 + mt * 16 + r + 8 * hseg;
        const int n = n0 + nt * 16 + lm;
        const size_t idx = ((size_t)b * CC + o) * NN + n;
        out[idx] = acc[mt][nt][r] + proj_b[o] + x[idx];
      }
}

// ---------------------------------------------------------------------------
extern "C" void kernel_launch(void* const* d_in, const int* in_sizes, int n_in,
                              void* d_out, int out_size, void* d_ws, size_t ws_size,
                              hipStream_t stream) {
  (void)in_sizes; (void)n_in; (void)out_size; (void)ws_size;
  const float* x      = (const float*)d_in[0];
  const float* gamma  = (const float*)d_in[1];
  const float* beta   = (const float*)d_in[2];
  const float* qkv_w  = (const float*)d_in[3];
  const float* qkv_b  = (const float*)d_in[4];
  const float* proj_w = (const float*)d_in[5];
  const float* proj_b = (const float*)d_in[6];
  float* out = (float*)d_out;

  // Workspace (f16): hT[B][N][C] (8MB, reused as attnO) | Q | K | Vt (8MB each)
  //                  | qkv_w16 (384KB) | proj_w16 (128KB)
  _Float16* hT   = (_Float16*)d_ws;
  _Float16* Q    = hT + (size_t)4 * NN * CC;
  _Float16* K    = Q  + (size_t)4 * NH * NN * HD;
  _Float16* Vt   = K  + (size_t)4 * NH * NN * HD;
  _Float16* wq16 = Vt + (size_t)4 * NH * NN * HD;
  _Float16* wp16 = wq16 + (size_t)3 * CC * CC;

  cvt_f32_to_f16<<<dim3((3 * CC * CC) / 1024), dim3(256), 0, stream>>>(qkv_w, wq16, 3 * CC * CC);
  cvt_f32_to_f16<<<dim3((CC * CC) / 1024), dim3(256), 0, stream>>>(proj_w, wp16, CC * CC);
  groupnorm_to_f16T<<<dim3(4 * 32), dim3(256), 0, stream>>>(x, gamma, beta, hT);
  qkv_gemm<<<dim3(NN / 128, 768 / 128, 4), dim3(256), 0, stream>>>(hT, wq16, qkv_b, Q, K, Vt);
  flash_attn<<<dim3(NN / 64, 4 * NH), dim3(128), 0, stream>>>(Q, K, Vt, hT);
  proj_gemm<<<dim3(NN / 128, CC / 128, 4), dim3(256), 0, stream>>>(hT, wp16, proj_b, x, out);
}
